// LrMSA_57062935495201
// MI455X (gfx1250) — matrix operand order repaired and checked by reading, not compiled, not run on hardware
//
#include <hip/hip_runtime.h>

// ---- CDNA5 WMMA types ----
typedef __attribute__((ext_vector_type(16))) __bf16 v16bf;
typedef __attribute__((ext_vector_type(8)))  float  v8f;

#define DIMF   768
#define HEADS  12
#define HD     64
#define RANK   8
#define SEQ    1024
#define BATCH  4
#define MTOT   (BATCH * SEQ)   /* 4096 */

// ---- CDNA5 async global->LDS copies (ASYNCcnt path), guarded ----
#if defined(__gfx1250__) && defined(__has_builtin)
#if __has_builtin(__builtin_amdgcn_global_load_async_to_lds_b128)
#define USE_ASYNC 1
#endif
#endif
#ifndef USE_ASYNC
#define USE_ASYNC 0
#endif

#if defined(__has_builtin) && __has_builtin(__builtin_amdgcn_s_wait_asynccnt)
#define WAIT_ASYNC(n) __builtin_amdgcn_s_wait_asynccnt(n)
#else
#define WAIT_ASYNC(n) asm volatile("s_wait_asynccnt %0" ::"i"(n) : "memory")
#endif

// Builtin signature (from clang diagnostic): params are
// 'int __attribute__((vector_size(16)))*' in __device__ (AS1) / __shared__ (AS3).
typedef int async_v4i __attribute__((vector_size(16)));
#define AS1P(p) ((__attribute__((address_space(1))) async_v4i*)(p))
#define AS3P(p) ((__attribute__((address_space(3))) async_v4i*)(p))

union Frag { v16bf v; uint4 u[2]; };

// A-fragment (16x32 bf16, ISA 7.12.2): lane L<16 -> row L, K 0..7 & 16..23;
// lane L+16 -> row L, K 8..15 & 24..31.  Source tile row-major, stride 32 elems.
__device__ inline v16bf fragA(const __bf16* t, int row0, int lane) {
  Frag f;
  const int row = row0 + (lane & 15);
  const int off = (lane < 16) ? 0 : 8;
  const uint4* p = reinterpret_cast<const uint4*>(t + row * 32 + off);
  f.u[0] = p[0];       // K off..off+7
  f.u[1] = p[2];       // K off+16..off+23
  return f.v;
}

// B-fragment (32x16 bf16): lane n<16 holds column n, K 0..7 & 16..23; lane n+16 holds K 8..15 & 24..31.
// Source tile is the *transposed* operand stored row-major [n][K], stride 32 elems.
__device__ inline v16bf fragB(const __bf16* t, int n0, int lane) {
  Frag f;
  const int n   = n0 + (lane & 15);
  const int off = (lane < 16) ? 0 : 8;
  const uint4* p = reinterpret_cast<const uint4*>(t + n * 32 + off);
  f.u[0] = p[0];
  f.u[1] = p[2];
  return f.v;
}

// ---------------- generic bf16 WMMA GEMM: C = X @ W^T + bias ----------------
// X [M,K] bf16 row-major, W [Nf,K] bf16 row-major, C [M,Nf] f32. 128x64 WG tile.
// Double-buffered LDS, async global->LDS staging, last K-tile peeled.
__global__ __launch_bounds__(256)
void gemm_xwt_bf16(const __bf16* __restrict__ X, const __bf16* __restrict__ W,
                   const float* __restrict__ bias, float* __restrict__ C,
                   int M, int Nf, int K) {
  const int tid  = threadIdx.x;
  const int lane = tid & 31;
  const int wave = tid >> 5;     // 0..7
  const int wm   = wave >> 1;    // 0..3 -> 32-row slab
  const int wn   = wave & 1;     // 0..1 -> 32-col slab
  const int mblk = blockIdx.x * 128;
  const int nblk = blockIdx.y * 64;

  v8f acc[2][2] = {};

  const int ar = tid >> 1, ah = (tid & 1) * 16;   // A: 128 rows x 2 half-rows
  const int br = tid >> 2, bc = (tid & 3) * 8;    // B: 64 rows x 4 chunks

#if USE_ASYNC
  __shared__ __bf16 As[2][128 * 32];
  __shared__ __bf16 Bs[2][64 * 32];

  auto issue = [&](int kt, int bufi) {
    const __bf16* ga = X + (size_t)(mblk + ar) * K + kt + ah;
    __bf16* la = &As[bufi][ar * 32 + ah];
    __builtin_amdgcn_global_load_async_to_lds_b128(AS1P(ga),     AS3P(la),     0, 0);
    __builtin_amdgcn_global_load_async_to_lds_b128(AS1P(ga + 8), AS3P(la + 8), 0, 0);
    const __bf16* gb = W + (size_t)(nblk + br) * K + kt + bc;
    __bf16* lb = &Bs[bufi][br * 32 + bc];
    __builtin_amdgcn_global_load_async_to_lds_b128(AS1P(gb), AS3P(lb), 0, 0);
  };
  auto compute = [&](const __bf16* A_, const __bf16* B_) {
    v16bf a0 = fragA(A_, wm * 32,      lane);
    v16bf a1 = fragA(A_, wm * 32 + 16, lane);
    v16bf b0 = fragB(B_, wn * 32,      lane);
    v16bf b1 = fragB(B_, wn * 32 + 16, lane);
    acc[0][0] = __builtin_amdgcn_wmma_f32_16x16x32_bf16(false, a0, false, b0, (short)0, acc[0][0], false, false);
    acc[0][1] = __builtin_amdgcn_wmma_f32_16x16x32_bf16(false, a0, false, b1, (short)0, acc[0][1], false, false);
    acc[1][0] = __builtin_amdgcn_wmma_f32_16x16x32_bf16(false, a1, false, b0, (short)0, acc[1][0], false, false);
    acc[1][1] = __builtin_amdgcn_wmma_f32_16x16x32_bf16(false, a1, false, b1, (short)0, acc[1][1], false, false);
  };

  issue(0, 0);                                   // prologue: tile 0 -> buf 0
  int t = 0;
  // steady state: always one tile in flight, unconditional waits
  for (int kt = 0; kt + 32 < K; kt += 32, ++t) {
    issue(kt + 32, (t + 1) & 1);                 // prefetch next tile into other buf
    WAIT_ASYNC(3);                               // current tile resident (in-order)
    __syncthreads();                             // visible to all waves
    compute(As[t & 1], Bs[t & 1]);
    __syncthreads();                             // protect buffer reuse
  }
  // epilogue: last tile
  WAIT_ASYNC(0);
  __syncthreads();
  compute(As[t & 1], Bs[t & 1]);
#else
  __shared__ __bf16 As[128 * 32];
  __shared__ __bf16 Bs[64 * 32];
  for (int kt = 0; kt < K; kt += 32) {
    __syncthreads();
    { // stage A tile 128x32
      const uint4* g = reinterpret_cast<const uint4*>(X + (size_t)(mblk + ar) * K + kt + ah);
      uint4* s = reinterpret_cast<uint4*>(As + ar * 32 + ah);
      s[0] = g[0]; s[1] = g[1];
    }
    { // stage B tile 64x32
      const uint4* g = reinterpret_cast<const uint4*>(W + (size_t)(nblk + br) * K + kt + bc);
      uint4* s = reinterpret_cast<uint4*>(Bs + br * 32 + bc);
      s[0] = g[0];
    }
    __syncthreads();

    v16bf a0 = fragA(As, wm * 32,      lane);
    v16bf a1 = fragA(As, wm * 32 + 16, lane);
    v16bf b0 = fragB(Bs, wn * 32,      lane);
    v16bf b1 = fragB(Bs, wn * 32 + 16, lane);
    acc[0][0] = __builtin_amdgcn_wmma_f32_16x16x32_bf16(false, a0, false, b0, (short)0, acc[0][0], false, false);
    acc[0][1] = __builtin_amdgcn_wmma_f32_16x16x32_bf16(false, a0, false, b1, (short)0, acc[0][1], false, false);
    acc[1][0] = __builtin_amdgcn_wmma_f32_16x16x32_bf16(false, a1, false, b0, (short)0, acc[1][0], false, false);
    acc[1][1] = __builtin_amdgcn_wmma_f32_16x16x32_bf16(false, a1, false, b1, (short)0, acc[1][1], false, false);
  }
#endif

  // C layout (ISA): VGPR r, lane<16 -> (M=r, N=lane); lane>=16 -> (M=r+8, N=lane-16)
  #pragma unroll
  for (int i = 0; i < 2; ++i)
    #pragma unroll
    for (int j = 0; j < 2; ++j) {
      const int n  = nblk + wn * 32 + j * 16 + (lane & 15);
      const float bv = bias[n];
      const int m0 = mblk + wm * 32 + i * 16 + ((lane < 16) ? 0 : 8);
      #pragma unroll
      for (int r = 0; r < 8; ++r)
        C[(size_t)(m0 + r) * Nf + n] = acc[i][j][r] + bv;
    }
}

// ---------------- helpers ----------------
// 4-wide f32 -> bf16 conversion (all sizes are multiples of 4)
__global__ void f32_to_bf16_v4(const float4* __restrict__ in, uint2* __restrict__ out, int n4) {
  int i = blockIdx.x * blockDim.x + threadIdx.x;
  if (i >= n4) return;
  const float4 v = in[i];
  union { __bf16 h[4]; uint2 u; } r;
  r.h[0] = (__bf16)v.x; r.h[1] = (__bf16)v.y;
  r.h[2] = (__bf16)v.z; r.h[3] = (__bf16)v.w;
  out[i] = r.u;
}

// xa[m,r] = sum_k x[m,k] * A[r,k]   (fp32, r=8 -> tiny)
__global__ void lora_xa(const float* __restrict__ x, const float* __restrict__ A,
                        float* __restrict__ xa) {
  int idx = blockIdx.x * blockDim.x + threadIdx.x;
  if (idx >= MTOT * RANK) return;
  const int m = idx >> 3, r = idx & 7;
  const float* xp = x + (size_t)m * DIMF;
  const float* ap = A + (size_t)r * DIMF;
  float s = 0.f;
  for (int k = 0; k < DIMF; ++k) s += xp[k] * ap[k];
  xa[idx] = s;
}

// q/k/v assembly: add qkv slice + base linear + LoRA delta, emit bf16 q,k [B,H,N,hd]
// and transposed v [B,H,hd,N] (so attention V B-fragments are contiguous loads).
__global__ void combine_qkv(const float* __restrict__ Cqkv, const float* __restrict__ Cqb,
                            const float* __restrict__ Ckb,  const float* __restrict__ xaq,
                            const float* __restrict__ xak,  const float* __restrict__ qB,
                            const float* __restrict__ kB,
                            __bf16* __restrict__ q, __bf16* __restrict__ k,
                            __bf16* __restrict__ vt) {
  int tid = blockIdx.x * blockDim.x + threadIdx.x;
  if (tid >= BATCH * HEADS * SEQ * HD) return;
  const int d = tid & 63;  int t = tid >> 6;
  const int n = t & 1023;  t >>= 10;
  const int h = t % HEADS; const int b = t / HEADS;
  const int m = b * SEQ + n;
  const int c = h * HD + d;
  float lq = 0.f, lk = 0.f;
  #pragma unroll
  for (int rr = 0; rr < RANK; ++rr) {
    lq += xaq[m * RANK + rr] * qB[c * RANK + rr];
    lk += xak[m * RANK + rr] * kB[c * RANK + rr];
  }
  const float qv = Cqkv[(size_t)m * 2304 + c]        + Cqb[(size_t)m * DIMF + c] + 0.125f * lq;
  const float kv = Cqkv[(size_t)m * 2304 + 768 + c]  + Ckb[(size_t)m * DIMF + c] + 0.125f * lk;
  const float vv = Cqkv[(size_t)m * 2304 + 1536 + c];
  const size_t bh = (size_t)(b * HEADS + h);
  q [(bh * SEQ + n) * HD + d] = (__bf16)qv;
  k [(bh * SEQ + n) * HD + d] = (__bf16)kv;
  vt[(bh * HD  + d) * SEQ + n] = (__bf16)vv;
}

// ---------------- flash attention, one wave per 16 query rows ----------------
// Q,K [BH,SEQ,HD] bf16; Vt [BH,HD,SEQ] bf16; out bf16 [MTOT, DIMF] (= [B,N,H*hd])
__global__ __launch_bounds__(128)
void attn_flash(const __bf16* __restrict__ Q, const __bf16* __restrict__ Km,
                const __bf16* __restrict__ Vt, __bf16* __restrict__ out) {
  __shared__ __bf16 Psh[4][16 * 32];
  const int lane = threadIdx.x & 31;
  const int wv   = threadIdx.x >> 5;              // 0..3
  const int bh   = blockIdx.x;                    // 0..47
  const int m0   = blockIdx.y * 64 + wv * 16;
  const int b = bh / HEADS, h = bh % HEADS;
  __bf16* P = Psh[wv];

  // Q A-fragments for hd 0..31 and 32..63, kept resident
  v16bf qf[2];
  {
    const __bf16* qp = Q + ((size_t)bh * SEQ + m0) * HD;
    const int row = lane & 15;
    const int off = (lane < 16) ? 0 : 8;
    #pragma unroll
    for (int ks = 0; ks < 2; ++ks) {
      Frag f;
      const uint4* p = reinterpret_cast<const uint4*>(qp + row * HD + ks * 32 + off);
      f.u[0] = p[0]; f.u[1] = p[2];
      qf[ks] = f.v;
    }
  }

  float mrow[8], lrow[8];
  v8f o[4] = {};
  #pragma unroll
  for (int r = 0; r < 8; ++r) { mrow[r] = -1e30f; lrow[r] = 0.f; }

  const __bf16* kb = Km + (size_t)bh * SEQ * HD;
  const __bf16* vb = Vt + (size_t)bh * HD * SEQ;

  for (int j = 0; j < SEQ; j += 32) {
    // S tiles: keys j..j+15 (s0) and j+16..j+31 (s1), K-dim = hd (2 steps of 32)
    v8f s0 = {}, s1 = {};
    #pragma unroll
    for (int ks = 0; ks < 2; ++ks) {
      Frag f0, f1;
      const int key = j + (lane & 15);
      const int hd0 = ks * 32 + ((lane < 16) ? 0 : 8);
      const uint4* p0 = reinterpret_cast<const uint4*>(kb + (size_t)key * HD + hd0);
      const uint4* p1 = reinterpret_cast<const uint4*>(kb + (size_t)(key + 16) * HD + hd0);
      f0.u[0] = p0[0]; f0.u[1] = p0[2];
      f1.u[0] = p1[0]; f1.u[1] = p1[2];
      s0 = __builtin_amdgcn_wmma_f32_16x16x32_bf16(false, qf[ks], false, f0.v, (short)0, s0, false, false);
      s1 = __builtin_amdgcn_wmma_f32_16x16x32_bf16(false, qf[ks], false, f1.v, (short)0, s1, false, false);
    }
    // online softmax; row r lives in lanes 0-15 (M=r) / 16-31 (M=r+8) of VGPR r
    #pragma unroll
    for (int r = 0; r < 8; ++r) {
      const float a = s0[r] * 0.125f;           // SCALE = hd^-0.5
      const float c = s1[r] * 0.125f;
      float mx = fmaxf(a, c);
      #pragma unroll
      for (int msk = 1; msk < 16; msk <<= 1) mx = fmaxf(mx, __shfl_xor(mx, msk, 32));
      const float mnew  = fmaxf(mrow[r], mx);
      const float alpha = __expf(mrow[r] - mnew);
      const float p0 = __expf(a - mnew);
      const float p1 = __expf(c - mnew);
      float rs = p0 + p1;
      #pragma unroll
      for (int msk = 1; msk < 16; msk <<= 1) rs += __shfl_xor(rs, msk, 32);
      lrow[r] = lrow[r] * alpha + rs;
      mrow[r] = mnew;
      o[0][r] *= alpha; o[1][r] *= alpha; o[2][r] *= alpha; o[3][r] *= alpha;
      // C-layout -> row-major P tile [16 rows x 32 keys] in LDS
      const int prow = r + ((lane < 16) ? 0 : 8);
      const int pcol = lane & 15;
      P[prow * 32 + pcol]      = (__bf16)p0;
      P[prow * 32 + 16 + pcol] = (__bf16)p1;
    }
    __syncthreads();
    Frag pf;   // P as A-fragment (16x32)
    {
      const int row = lane & 15;
      const int off = (lane < 16) ? 0 : 8;
      const uint4* p = reinterpret_cast<const uint4*>(P + row * 32 + off);
      pf.u[0] = p[0]; pf.u[1] = p[2];
    }
    #pragma unroll
    for (int t = 0; t < 4; ++t) {   // O += P @ V, hd columns t*16..t*16+15
      Frag vf;
      const int hd   = t * 16 + (lane & 15);
      const int koff = j + ((lane < 16) ? 0 : 8);
      const uint4* p = reinterpret_cast<const uint4*>(vb + (size_t)hd * SEQ + koff);
      vf.u[0] = p[0]; vf.u[1] = p[2];
      o[t] = __builtin_amdgcn_wmma_f32_16x16x32_bf16(false, pf.v, false, vf.v, (short)0, o[t], false, false);
    }
    __syncthreads();
  }

  const int rg0 = b * SEQ + m0 + ((lane < 16) ? 0 : 8);
  #pragma unroll
  for (int t = 0; t < 4; ++t) {
    const int col = h * HD + t * 16 + (lane & 15);
    #pragma unroll
    for (int r = 0; r < 8; ++r)
      out[(size_t)(rg0 + r) * DIMF + col] = (__bf16)(o[t][r] / lrow[r]);
  }
}

// ---------------- launch ----------------
extern "C" void kernel_launch(void* const* d_in, const int* in_sizes, int n_in,
                              void* d_out, int out_size, void* d_ws, size_t ws_size,
                              hipStream_t stream) {
  const float* x      = (const float*)d_in[0];
  const float* qkv_b  = (const float*)d_in[2];
  const float* qb_b   = (const float*)d_in[4];
  const float* q_A    = (const float*)d_in[5];
  const float* q_B    = (const float*)d_in[6];
  const float* kb_b   = (const float*)d_in[8];
  const float* k_A    = (const float*)d_in[9];
  const float* k_B    = (const float*)d_in[10];
  const float* proj_b = (const float*)d_in[12];
  float* out = (float*)d_out;

  char* w = (char*)d_ws;
  size_t off = 0;
  auto alloc = [&](size_t bytes) -> void* {
    void* p = w + off;
    off = (off + bytes + 255) & ~(size_t)255;
    return p;
  };
  __bf16* x_bf    = (__bf16*)alloc((size_t)MTOT * DIMF * 2);
  __bf16* wqkv_bf = (__bf16*)alloc((size_t)2304 * DIMF * 2);
  __bf16* wq_bf   = (__bf16*)alloc((size_t)DIMF * DIMF * 2);
  __bf16* wk_bf   = (__bf16*)alloc((size_t)DIMF * DIMF * 2);
  __bf16* wp_bf   = (__bf16*)alloc((size_t)DIMF * DIMF * 2);
  float*  Cqkv    = (float*) alloc((size_t)MTOT * 2304 * 4);
  float*  Cqb     = (float*) alloc((size_t)MTOT * DIMF * 4);
  float*  Ckb     = (float*) alloc((size_t)MTOT * DIMF * 4);
  float*  xaq     = (float*) alloc((size_t)MTOT * RANK * 4);
  float*  xak     = (float*) alloc((size_t)MTOT * RANK * 4);
  __bf16* q_bf    = (__bf16*)alloc((size_t)MTOT * DIMF * 2);   // [B,H,N,hd]
  __bf16* k_bf    = (__bf16*)alloc((size_t)MTOT * DIMF * 2);   // [B,H,N,hd]
  __bf16* vt_bf   = (__bf16*)alloc((size_t)MTOT * DIMF * 2);   // [B,H,hd,N]
  __bf16* at_bf   = (__bf16*)alloc((size_t)MTOT * DIMF * 2);   // [B*N, D]

  auto cvt = [&](const float* src, __bf16* dst, int n) {
    const int n4 = n / 4;
    f32_to_bf16_v4<<<(n4 + 255) / 256, 256, 0, stream>>>(
        (const float4*)src, (uint2*)dst, n4);
  };
  cvt(x,                        x_bf,    MTOT * DIMF);
  cvt((const float*)d_in[1],    wqkv_bf, 2304 * DIMF);   // qkv_w
  cvt((const float*)d_in[3],    wq_bf,   DIMF * DIMF);   // q_base_w
  cvt((const float*)d_in[7],    wk_bf,   DIMF * DIMF);   // k_base_w
  cvt((const float*)d_in[11],   wp_bf,   DIMF * DIMF);   // proj_w

  lora_xa<<<(MTOT * RANK + 255) / 256, 256, 0, stream>>>(x, q_A, xaq);
  lora_xa<<<(MTOT * RANK + 255) / 256, 256, 0, stream>>>(x, k_A, xak);

  gemm_xwt_bf16<<<dim3(MTOT / 128, 2304 / 64), 256, 0, stream>>>(x_bf, wqkv_bf, qkv_b, Cqkv, MTOT, 2304, DIMF);
  gemm_xwt_bf16<<<dim3(MTOT / 128, DIMF / 64), 256, 0, stream>>>(x_bf, wq_bf,   qb_b,  Cqb,  MTOT, DIMF, DIMF);
  gemm_xwt_bf16<<<dim3(MTOT / 128, DIMF / 64), 256, 0, stream>>>(x_bf, wk_bf,   kb_b,  Ckb,  MTOT, DIMF, DIMF);

  combine_qkv<<<(BATCH * HEADS * SEQ * HD + 255) / 256, 256, 0, stream>>>(
      Cqkv, Cqb, Ckb, xaq, xak, q_B, k_B, q_bf, k_bf, vt_bf);

  attn_flash<<<dim3(BATCH * HEADS, SEQ / 64), 128, 0, stream>>>(q_bf, k_bf, vt_bf, at_bf);

  gemm_xwt_bf16<<<dim3(MTOT / 128, DIMF / 64), 256, 0, stream>>>(at_bf, wp_bf, proj_b, out, MTOT, DIMF, DIMF);
}
